// Transformer_73091753443702
// MI455X (gfx1250) — matrix operand order repaired and checked by reading, not compiled, not run on hardware
//
#include <hip/hip_runtime.h>
#include <hip/hip_bf16.h>

#define D_MODEL 512
#define HID_DIM 2048
#define NHEAD   8
#define DHEAD   64
#define NLAYER  6
#define BATCH   8
#define SEQ     512
#define EPS_LN  1e-5f

typedef __attribute__((ext_vector_type(16))) _Float16 v16h;
typedef __attribute__((ext_vector_type(8)))  _Float16 v8h;
typedef __attribute__((ext_vector_type(8)))  float    v8f;

// ---------------------------------------------------------------------------
// A-fragment loader. 16-bit A 16x32 (wave32), contiguous k order used here (B matches):
//   lane half hf: halfs 0..7  <- K = k0 + 16*hf + 0..7     (16B contiguous)
//                 halfs 8..15 <- K = k0 + 16*hf + 8 + 0..7 (16B contiguous)
// ---------------------------------------------------------------------------
__device__ __forceinline__ v16h load_a_frag(const _Float16* __restrict__ row,
                                            int k0, int hf)
{
    const v8h lo = *reinterpret_cast<const v8h*>(row + k0 + 16 * hf);
    const v8h hi = *reinterpret_cast<const v8h*>(row + k0 + 16 * hf + 8);
    v16h a;
#pragma unroll
    for (int i = 0; i < 8; ++i) { a[i] = lo[i]; a[8 + i] = hi[i]; }
    return a;
}

// ---------------------------------------------------------------------------
// Batched GEMM on v_wmma_f32_16x16x32_f16.
//   C[M,N] = act(A[M,K] * B^T + bias), A f16 row-major (lda), B f16 stored
//   transposed as Bt[N,K] row-major (ldb). Output f32 (Cf) or f16 (Ch).
// Each wave computes a 64x64 tile: 4 A frags x 4 B frags -> 16 WMMAs/k-step
// against 16 b128 loads. B fragments are software-pipelined: fragment j+1's
// loads issue before the 4 WMMAs consuming fragment j, overlapping
// s_wait_loadcnt with matrix math.
//   B 32x16 frag: lane half hf: halfs 0..15 <- K = k0 + 16*hf + 0..15 (32B)
//   C 16x16:      VGPR j -> row hf*8+j, col (lane&15)
// ---------------------------------------------------------------------------
__global__ __launch_bounds__(256) void wmma_gemm_f16_kernel(
    const _Float16* __restrict__ A, const _Float16* __restrict__ Bt,
    const float* __restrict__ bias, float* __restrict__ Cf,
    _Float16* __restrict__ Ch,
    int M, int N, int K, int lda, int ldb, int ldc,
    int nh, int aSB, int aSH, int bSB, int bSH, int cSB, int cSH, int relu)
{
    const int gwave     = blockIdx.x * (blockDim.x >> 5) + (threadIdx.x >> 5);
    const int tilesN    = N >> 6;
    const int waveTiles = (M >> 6) * tilesN;
    if (gwave >= waveTiles) return;          // uniform whole-wave exit

    const int tn   = gwave % tilesN;
    const int tm   = gwave / tilesN;
    const int lane = threadIdx.x & 31;
    const int r    = lane & 15;
    const int hf   = lane >> 4;

    const int zb = blockIdx.z / nh, zh = blockIdx.z % nh;
    const _Float16* Ab = A  + (size_t)zb * aSB + (size_t)zh * aSH;
    const _Float16* Bb = Bt + (size_t)zb * bSB + (size_t)zh * bSH;

    const _Float16* Arow0 = Ab + (size_t)(tm * 64 + r) * lda;
    const size_t    aRow16 = (size_t)16 * lda;

    v8f acc[16];
#pragma unroll
    for (int t = 0; t < 16; ++t)
#pragma unroll
        for (int i = 0; i < 8; ++i) acc[t][i] = 0.0f;

    for (int k0 = 0; k0 < K; k0 += 32) {
        const v16h a0 = load_a_frag(Arow0,              k0, hf);
        const v16h a1 = load_a_frag(Arow0 + 1 * aRow16, k0, hf);
        const v16h a2 = load_a_frag(Arow0 + 2 * aRow16, k0, hf);
        const v16h a3 = load_a_frag(Arow0 + 3 * aRow16, k0, hf);
        if (k0 + 32 < K) {
            __builtin_prefetch(Arow0 + 0 * aRow16 + k0 + 32, 0, 3);
            __builtin_prefetch(Arow0 + 1 * aRow16 + k0 + 32, 0, 3);
            __builtin_prefetch(Arow0 + 2 * aRow16 + k0 + 32, 0, 3);
            __builtin_prefetch(Arow0 + 3 * aRow16 + k0 + 32, 0, 3);
        }
        const _Float16* BrowBase = Bb + (size_t)(tn * 64 + r) * ldb
                                      + k0 + 16 * hf;
        v16h bcur = *reinterpret_cast<const v16h*>(BrowBase);
#pragma unroll
        for (int j = 0; j < 4; ++j) {
            v16h bnxt = bcur;
            if (j < 3)
                bnxt = *reinterpret_cast<const v16h*>(
                    BrowBase + (size_t)(j + 1) * 16 * ldb);
            acc[j] = __builtin_amdgcn_wmma_f32_16x16x32_f16(
                false, a0, false, bcur, (short)0, acc[j], false, false);
            acc[4 + j] = __builtin_amdgcn_wmma_f32_16x16x32_f16(
                false, a1, false, bcur, (short)0, acc[4 + j], false, false);
            acc[8 + j] = __builtin_amdgcn_wmma_f32_16x16x32_f16(
                false, a2, false, bcur, (short)0, acc[8 + j], false, false);
            acc[12 + j] = __builtin_amdgcn_wmma_f32_16x16x32_f16(
                false, a3, false, bcur, (short)0, acc[12 + j], false, false);
            bcur = bnxt;
        }
    }

#pragma unroll
    for (int m = 0; m < 4; ++m)
#pragma unroll
        for (int j = 0; j < 4; ++j) {
            const int colg = tn * 64 + j * 16 + r;
            const float bv = bias ? bias[colg] : 0.0f;
#pragma unroll
            for (int i = 0; i < 8; ++i) {
                const int rowg = tm * 64 + m * 16 + hf * 8 + i;
                float v = acc[m * 4 + j][i] + bv;
                if (relu) v = fmaxf(v, 0.0f);
                if (Ch) Ch[(size_t)rowg * ldc + colg] = (_Float16)v;
                else    Cf[(size_t)rowg * ldc + colg] = v;
            }
        }
}

// ---------------------------------------------------------------------------
// LDS-tiled transpose + f32->f16 cast:  out[n*ldout + k] = in[k*ldin + n]
// ---------------------------------------------------------------------------
__global__ __launch_bounds__(256) void transpose_cast_kernel(
    const float* __restrict__ in, _Float16* __restrict__ out,
    int ldin, int ldout, int nh, int inSB, int inSH, int outSB, int outSH)
{
    __shared__ float tile[32][33];
    const int zb = blockIdx.z / nh, zh = blockIdx.z % nh;
    const float* inb  = in  + (size_t)zb * inSB  + (size_t)zh * inSH;
    _Float16*    outb = out + (size_t)zb * outSB + (size_t)zh * outSH;
    const int k0 = blockIdx.x * 32, n0 = blockIdx.y * 32;
    const int tx = threadIdx.x, ty = threadIdx.y;       // 32 x 8
#pragma unroll
    for (int i = 0; i < 4; ++i)
        tile[ty + 8 * i][tx] = inb[(size_t)(k0 + ty + 8 * i) * ldin + n0 + tx];
    __syncthreads();
#pragma unroll
    for (int i = 0; i < 4; ++i)
        outb[(size_t)(n0 + ty + 8 * i) * ldout + k0 + tx] =
            (_Float16)tile[tx][ty + 8 * i];
}

// ---------------------------------------------------------------------------
// wave32 reductions
// ---------------------------------------------------------------------------
__device__ __forceinline__ float wave_max(float v) {
#pragma unroll
    for (int m = 16; m >= 1; m >>= 1) v = fmaxf(v, __shfl_xor(v, m, 32));
    return v;
}
__device__ __forceinline__ float wave_sum(float v) {
#pragma unroll
    for (int m = 16; m >= 1; m >>= 1) v += __shfl_xor(v, m, 32);
    return v;
}

// ---------------------------------------------------------------------------
// Masked softmax, in-place on f16 scores; one wave per row (Sk=512).
// ---------------------------------------------------------------------------
__global__ __launch_bounds__(256) void softmax_mask_kernel(
    _Float16* __restrict__ scores, const int* __restrict__ qtok,
    const int* __restrict__ ktok, int rows, int Sq, int Sk,
    int causal, float scale)
{
    const int wid = blockIdx.x * (blockDim.x >> 5) + (threadIdx.x >> 5);
    if (wid >= rows) return;
    const int lane = threadIdx.x & 31;
    const int q = wid % Sq;
    const int b = (wid / Sq) / NHEAD;
    _Float16* row = scores + (size_t)wid * Sk;
    const bool qpad = (qtok[b * Sq + q] == 0);
    const float NEG = -__builtin_inff();

    float vals[16];
    float mx = NEG;
    const int npl = Sk >> 5;
    for (int t = 0; t < npl; ++t) {
        const int k = lane + (t << 5);
        const bool m = qpad || (ktok[b * Sk + k] == 0) || (causal && (k > q));
        const float v = m ? NEG : (float)row[k] * scale;
        vals[t] = v;
        mx = fmaxf(mx, v);
    }
    mx = wave_max(mx);
    float sum = 0.0f;
    for (int t = 0; t < npl; ++t) {
        const float e = (vals[t] == NEG) ? 0.0f : __expf(vals[t] - mx);
        vals[t] = e;
        sum += e;
    }
    sum = wave_sum(sum);
    const float inv = (sum > 0.0f) ? (1.0f / sum) : 0.0f;
    for (int t = 0; t < npl; ++t)
        row[lane + (t << 5)] = (_Float16)(vals[t] * inv);
}

// ---------------------------------------------------------------------------
// out = LayerNorm(x + r)*gamma + beta, plus f16 shadow; one wave per row.
// ---------------------------------------------------------------------------
__global__ __launch_bounds__(256) void add_layernorm_kernel(
    float* __restrict__ out, _Float16* __restrict__ out16,
    const float* __restrict__ x, const float* __restrict__ r,
    const float* __restrict__ gamma, const float* __restrict__ beta, int rows)
{
    const int wid = blockIdx.x * (blockDim.x >> 5) + (threadIdx.x >> 5);
    if (wid >= rows) return;
    const int lane = threadIdx.x & 31;
    const float* xr = x + (size_t)wid * D_MODEL;
    const float* rr = r + (size_t)wid * D_MODEL;

    float v[16];
    float s = 0.0f;
#pragma unroll
    for (int t = 0; t < 16; ++t) {
        const int d = lane + (t << 5);
        v[t] = xr[d] + rr[d];
        s += v[t];
    }
    s = wave_sum(s);
    const float mu = s * (1.0f / (float)D_MODEL);
    float s2 = 0.0f;
#pragma unroll
    for (int t = 0; t < 16; ++t) { const float dv = v[t] - mu; s2 += dv * dv; }
    s2 = wave_sum(s2);
    const float rs = rsqrtf(s2 * (1.0f / (float)D_MODEL) + EPS_LN);

    float*    of = out   + (size_t)wid * D_MODEL;
    _Float16* oh = out16 + (size_t)wid * D_MODEL;
#pragma unroll
    for (int t = 0; t < 16; ++t) {
        const int d = lane + (t << 5);
        const float o = (v[t] - mu) * rs * gamma[d] + beta[d];
        of[d] = o;
        oh[d] = (_Float16)o;
    }
}

// ---------------------------------------------------------------------------
// x[b,s,:] = em[token] + sinusoidal PE(s), f32 + f16 shadow
// ---------------------------------------------------------------------------
__global__ __launch_bounds__(256) void embed_pe_kernel(
    float* __restrict__ out, _Float16* __restrict__ out16,
    const float* __restrict__ em, const int* __restrict__ tok, int n)
{
    const int idx = blockIdx.x * blockDim.x + threadIdx.x;
    if (idx >= n) return;
    const int d   = idx & (D_MODEL - 1);
    const int row = idx >> 9;
    const int s   = row & (SEQ - 1);
    const float e = em[(size_t)tok[row] * D_MODEL + d];
    const float kFactor = -1.7988946e-2f;          // -ln(10000)/512
    const float ang = (float)s * __expf((float)(d & ~1) * kFactor);
    const float pe = (d & 1) ? __cosf(ang) : __sinf(ang);
    const float o = e + pe;
    out[idx]   = o;
    out16[idx] = (_Float16)o;
}

// ---------------------------------------------------------------------------
// host helpers
// ---------------------------------------------------------------------------
static void launch_gemm(hipStream_t st, const _Float16* A, const _Float16* Bt,
                        const float* bias, float* Cf, _Float16* Ch,
                        int M, int N, int K, int lda, int ldb, int ldc,
                        int batch, int nh, int aSB, int aSH, int bSB, int bSH,
                        int cSB, int cSH, int relu)
{
    const int waveTiles = (M >> 6) * (N >> 6);
    dim3 grid((waveTiles + 7) / 8, 1, batch);
    wmma_gemm_f16_kernel<<<grid, 256, 0, st>>>(A, Bt, bias, Cf, Ch, M, N, K,
                                               lda, ldb, ldc, nh, aSB, aSH,
                                               bSB, bSH, cSB, cSH, relu);
}

static void launch_transpose(hipStream_t st, const float* in, _Float16* out,
                             int K, int N, int ldin, int ldout, int batch,
                             int nh, int inSB, int inSH, int outSB, int outSH)
{
    dim3 grid(K / 32, N / 32, batch);
    transpose_cast_kernel<<<grid, dim3(32, 8), 0, st>>>(
        in, out, ldin, ldout, nh, inSB, inSH, outSB, outSH);
}

static void launch_ln(hipStream_t st, float* out, _Float16* out16,
                      const float* x, const float* r, const float* g,
                      const float* b, int rows)
{
    add_layernorm_kernel<<<(rows + 7) / 8, 256, 0, st>>>(out, out16, x, r, g,
                                                         b, rows);
}

// MHA: weight transpose-cast -> QKV proj -> V transpose -> QK^T -> softmax
//      -> PV -> out-proj(residual into t1)
static void run_mha(hipStream_t st, const _Float16* xq16, const _Float16* xkv16,
                    const int* qtok, const int* ktok, int causal,
                    const float* w, const float* bvec,
                    _Float16* wbuf, _Float16* q16, _Float16* k16, float* vb,
                    _Float16* vT16, _Float16* ob16, _Float16* sc16, float* t1)
{
    const int M  = BATCH * SEQ;
    const int DD = D_MODEL * D_MODEL;
    launch_transpose(st, w, wbuf, D_MODEL, D_MODEL, D_MODEL, D_MODEL,
                     4, 1, DD, 0, DD, 0);
    launch_gemm(st, xq16,  wbuf,          bvec,               nullptr, q16,
                M, D_MODEL, D_MODEL, D_MODEL, D_MODEL, D_MODEL,
                1, 1, 0,0,0,0,0,0, 0);
    launch_gemm(st, xkv16, wbuf + 1 * DD, bvec + 1 * D_MODEL, nullptr, k16,
                M, D_MODEL, D_MODEL, D_MODEL, D_MODEL, D_MODEL,
                1, 1, 0,0,0,0,0,0, 0);
    launch_gemm(st, xkv16, wbuf + 2 * DD, bvec + 2 * D_MODEL, vb, nullptr,
                M, D_MODEL, D_MODEL, D_MODEL, D_MODEL, D_MODEL,
                1, 1, 0,0,0,0,0,0, 0);
    launch_transpose(st, vb, vT16, SEQ, DHEAD, D_MODEL, SEQ,
                     BATCH * NHEAD, NHEAD,
                     SEQ * D_MODEL, DHEAD, NHEAD * DHEAD * SEQ, DHEAD * SEQ);
    launch_gemm(st, q16, k16, nullptr, nullptr, sc16, SEQ, SEQ, DHEAD,
                D_MODEL, D_MODEL, SEQ,
                BATCH * NHEAD, NHEAD,
                SEQ * D_MODEL, DHEAD, SEQ * D_MODEL, DHEAD,
                NHEAD * SEQ * SEQ, SEQ * SEQ, 0);
    const int rows = BATCH * NHEAD * SEQ;
    softmax_mask_kernel<<<(rows + 7) / 8, 256, 0, st>>>(
        sc16, qtok, ktok, rows, SEQ, SEQ, causal, 0.125f);
    launch_gemm(st, sc16, vT16, nullptr, nullptr, ob16, SEQ, DHEAD, SEQ,
                SEQ, SEQ, D_MODEL,
                BATCH * NHEAD, NHEAD,
                NHEAD * SEQ * SEQ, SEQ * SEQ,
                NHEAD * DHEAD * SEQ, DHEAD * SEQ,
                SEQ * D_MODEL, DHEAD, 0);
    launch_gemm(st, ob16, wbuf + 3 * DD, bvec + 3 * D_MODEL, t1, nullptr,
                M, D_MODEL, D_MODEL, D_MODEL, D_MODEL, D_MODEL,
                1, 1, 0,0,0,0,0,0, 0);
}

static void run_ffn(hipStream_t st, const _Float16* x16, const float* w1,
                    const float* b1, const float* w2, const float* b2,
                    _Float16* wbuf, _Float16* hid16, float* t1)
{
    const int M = BATCH * SEQ;
    launch_transpose(st, w1, wbuf, D_MODEL, HID_DIM, HID_DIM, D_MODEL,
                     1, 1, 0, 0, 0, 0);
    launch_transpose(st, w2, wbuf + D_MODEL * HID_DIM, HID_DIM, D_MODEL,
                     D_MODEL, HID_DIM, 1, 1, 0, 0, 0, 0);
    launch_gemm(st, x16, wbuf, b1, nullptr, hid16, M, HID_DIM, D_MODEL,
                D_MODEL, D_MODEL, HID_DIM, 1, 1, 0,0,0,0,0,0, 1);   // ReLU
    launch_gemm(st, hid16, wbuf + D_MODEL * HID_DIM, b2, t1, nullptr,
                M, D_MODEL, HID_DIM, HID_DIM, HID_DIM, D_MODEL,
                1, 1, 0,0,0,0,0,0, 0);
}

// ---------------------------------------------------------------------------
extern "C" void kernel_launch(void* const* d_in, const int* in_sizes, int n_in,
                              void* d_out, int out_size, void* d_ws,
                              size_t ws_size, hipStream_t stream)
{
    (void)in_sizes; (void)n_in; (void)out_size; (void)ws_size;

    const int*   inputs     = (const int*)d_in[0];
    const int*   outputs    = (const int*)d_in[1];
    const float* emi        = (const float*)d_in[2];
    const float* emo        = (const float*)d_in[3];
    const float* enc_attn_w = (const float*)d_in[4];
    const float* enc_attn_b = (const float*)d_in[5];
    const float* enc_ln     = (const float*)d_in[6];
    const float* enc_ffn_w1 = (const float*)d_in[7];
    const float* enc_ffn_b1 = (const float*)d_in[8];
    const float* enc_ffn_w2 = (const float*)d_in[9];
    const float* enc_ffn_b2 = (const float*)d_in[10];
    const float* dec_self_w = (const float*)d_in[11];
    const float* dec_self_b = (const float*)d_in[12];
    const float* dec_cross_w= (const float*)d_in[13];
    const float* dec_cross_b= (const float*)d_in[14];
    const float* dec_ln     = (const float*)d_in[15];
    const float* dec_ffn_w1 = (const float*)d_in[16];
    const float* dec_ffn_b1 = (const float*)d_in[17];
    const float* dec_ffn_w2 = (const float*)d_in[18];
    const float* dec_ffn_b2 = (const float*)d_in[19];

    // ---- workspace carve-up ----
    const size_t ND = (size_t)BATCH * SEQ * D_MODEL;      // 2M elements
    float* fp = (float*)d_ws;
    float* x  = fp; fp += ND;
    float* y  = fp; fp += ND;
    float* vb = fp; fp += ND;           // aliased as t1 (disjoint lifetimes)
    float* t1 = vb;
    _Float16* hp = (_Float16*)fp;
    _Float16* x16  = hp; hp += ND;
    _Float16* y16  = hp; hp += ND;
    _Float16* q16  = hp; hp += ND;
    _Float16* k16  = hp; hp += ND;
    _Float16* vT16 = hp; hp += ND;
    _Float16* ob16 = hp; hp += ND;
    _Float16* hid16= hp; hp += (size_t)BATCH * SEQ * HID_DIM;
    _Float16* sc16 = hp; hp += (size_t)BATCH * NHEAD * SEQ * SEQ;
    _Float16* wbuf = hp;                // 2*D*HID halfs staging slab

    const int M = BATCH * SEQ;
    const int n = (int)ND;

    embed_pe_kernel<<<(n + 255) / 256, 256, 0, stream>>>(x, x16, emi, inputs, n);
    embed_pe_kernel<<<(n + 255) / 256, 256, 0, stream>>>(y, y16, emo, outputs, n);

    // ---------------- encoder ----------------
    for (int l = 0; l < NLAYER; ++l) {
        const float* aw = enc_attn_w + (size_t)l * 4 * D_MODEL * D_MODEL;
        const float* ab = enc_attn_b + (size_t)l * 4 * D_MODEL;
        run_mha(stream, x16, x16, inputs, inputs, 0, aw, ab,
                wbuf, q16, k16, vb, vT16, ob16, sc16, t1);
        launch_ln(stream, x, x16, x, t1,
                  enc_ln + ((size_t)l * 2 + 0) * 2 * D_MODEL,
                  enc_ln + (((size_t)l * 2 + 0) * 2 + 1) * D_MODEL, M);
        run_ffn(stream, x16,
                enc_ffn_w1 + (size_t)l * D_MODEL * HID_DIM,
                enc_ffn_b1 + (size_t)l * HID_DIM,
                enc_ffn_w2 + (size_t)l * HID_DIM * D_MODEL,
                enc_ffn_b2 + (size_t)l * D_MODEL, wbuf, hid16, t1);
        launch_ln(stream, x, x16, x, t1,
                  enc_ln + ((size_t)l * 2 + 1) * 2 * D_MODEL,
                  enc_ln + (((size_t)l * 2 + 1) * 2 + 1) * D_MODEL, M);
    }

    // ---------------- decoder ----------------
    for (int l = 0; l < NLAYER; ++l) {
        const float* sw = dec_self_w  + (size_t)l * 4 * D_MODEL * D_MODEL;
        const float* sb = dec_self_b  + (size_t)l * 4 * D_MODEL;
        const float* cw = dec_cross_w + (size_t)l * 4 * D_MODEL * D_MODEL;
        const float* cb = dec_cross_b + (size_t)l * 4 * D_MODEL;

        run_mha(stream, y16, y16, outputs, outputs, 1, sw, sb,
                wbuf, q16, k16, vb, vT16, ob16, sc16, t1);
        launch_ln(stream, y, y16, y, t1,
                  dec_ln + ((size_t)l * 3 + 0) * 2 * D_MODEL,
                  dec_ln + (((size_t)l * 3 + 0) * 2 + 1) * D_MODEL, M);

        run_mha(stream, y16, x16, outputs, inputs, 0, cw, cb,
                wbuf, q16, k16, vb, vT16, ob16, sc16, t1);
        launch_ln(stream, y, y16, y, t1,
                  dec_ln + ((size_t)l * 3 + 1) * 2 * D_MODEL,
                  dec_ln + (((size_t)l * 3 + 1) * 2 + 1) * D_MODEL, M);

        run_ffn(stream, y16,
                dec_ffn_w1 + (size_t)l * D_MODEL * HID_DIM,
                dec_ffn_b1 + (size_t)l * HID_DIM,
                dec_ffn_w2 + (size_t)l * HID_DIM * D_MODEL,
                dec_ffn_b2 + (size_t)l * D_MODEL, wbuf, hid16, t1);
        float* dst = (l == NLAYER - 1) ? (float*)d_out : y;
        launch_ln(stream, dst, y16, y, t1,
                  dec_ln + ((size_t)l * 3 + 2) * 2 * D_MODEL,
                  dec_ln + (((size_t)l * 3 + 2) * 2 + 1) * D_MODEL, M);
    }
}
